// PainnUpdate_23313082483621
// MI455X (gfx1250) — matrix operand order repaired and checked by reading, not compiled, not run on hardware
//
#include <hip/hip_runtime.h>
#include <hip/hip_bf16.h>

typedef __attribute__((ext_vector_type(16))) __bf16 v16bf;
typedef __attribute__((ext_vector_type(8)))  float  v8f;

#define WAVES 4
#define EPSV 1e-6f
#define E2V  1e-12f

// d_ws layout (bf16 element offsets) — all transposed Bt[N][K], scales folded in
#define OFF_WU0   0        // 128x128
#define OFF_WV0   16384    // 128x128
#define OFF_WU1   32768    // 64x64
#define OFF_WV1   36864    // 64x64
#define OFF_WU2   40960    // 32x32
#define OFF_WV2   41984    // 32x32
#define OFF_WDOT  43008    // 128x224
#define OFF_WM1   71680    // 128x352
#define OFF_WM2   116736   // 480x128

// per-wave LDS (bf16 elems): sU 7680 | sDots 3584 | sInv 3584 | sH 2048 | sInner 2048
#define LDS_PER_WAVE 18944

__global__ void prep_transpose(const float* __restrict__ src, __bf16* __restrict__ dst,
                               int K, int Ncols, float scale) {
  int idx = blockIdx.x * blockDim.x + threadIdx.x;
  int total = K * Ncols;
  if (idx >= total) return;
  int n = idx / K;
  int k = idx - n * K;
  dst[idx] = (__bf16)(src[k * Ncols + n] * scale);
}

__device__ __forceinline__ v8f wmma_bf(v16bf a, v16bf b, v8f c) {
  return __builtin_amdgcn_wmma_f32_16x16x32_bf16(false, a, false, b, (short)0, c, false, false);
}

// A fragment (16x32) from global f32, row-major, row stride ld
__device__ __forceinline__ v16bf loadA_g(const float* __restrict__ base, int ld,
                                         int n0, int k0, int lane) {
  int r  = lane & 15;
  int kb = k0 + ((lane >> 4) << 3);
  const float* p = base + (size_t)(n0 + r) * ld + kb;
  v16bf a;
#pragma unroll
  for (int i = 0; i < 8; ++i) { a[i] = (__bf16)p[i]; a[i + 8] = (__bf16)p[i + 16]; }
  return a;
}

// A fragment from interleaved irrep storage: x[n, u, m] = xp[n*480 + cbase + u*stride + m]
__device__ __forceinline__ v16bf loadA_g_strided(const float* __restrict__ xp, int n0,
                                                 int cbase, int stride, int m, int k0, int lane) {
  int r  = lane & 15;
  int kb = k0 + ((lane >> 4) << 3);
  const float* p = xp + (size_t)(n0 + r) * 480 + cbase + m;
  v16bf a;
#pragma unroll
  for (int i = 0; i < 8; ++i) {
    a[i]     = (__bf16)p[(kb + i) * stride];
    a[i + 8] = (__bf16)p[(kb + i + 16) * stride];
  }
  return a;
}

// A fragment from LDS bf16 row-major (16 rows x ld cols)
__device__ __forceinline__ v16bf loadA_lds(const __bf16* S, int ld, int k0, int lane) {
  int r  = lane & 15;
  int kb = k0 + ((lane >> 4) << 3);
  const __bf16* p = S + r * ld + kb;
  v16bf a;
#pragma unroll
  for (int i = 0; i < 8; ++i) { a[i] = p[i]; a[i + 8] = p[i + 16]; }
  return a;
}

// B fragment (32x16) from transposed weights Bt[N][K] (bf16, row-major in K)
__device__ __forceinline__ v16bf loadB(const __bf16* __restrict__ Bt, int K,
                                       int col0, int k0, int lane) {
  const __bf16* p = Bt + (size_t)(col0 + (lane & 15)) * K + k0 + ((lane >> 4) << 3);
  v16bf b;
#pragma unroll
  for (int i = 0; i < 8; ++i) { b[i] = p[i]; b[i + 8] = p[i + 16]; }
  return b;
}

// D-tile <-> LDS in per-lane linear layout (lossless mapping, bf16)
__device__ __forceinline__ void storeTile(__bf16* T, const v8f& acc, int lane) {
  __bf16* p = T + lane * 8;
#pragma unroll
  for (int i = 0; i < 8; ++i) p[i] = (__bf16)acc[i];
}
__device__ __forceinline__ void loadTile(const __bf16* T, float* u, int lane) {
  const __bf16* p = T + lane * 8;
#pragma unroll
  for (int i = 0; i < 8; ++i) u[i] = (float)p[i];
}

// D-tile -> LDS row-major (for later A-fragment reads). D layout: lane -> col=lane&15,
// rows = i + 8*(lane>>4)
__device__ __forceinline__ void storeRowMajor(__bf16* S, int ld, int col0,
                                              const v8f& vals, int lane) {
  int c  = col0 + (lane & 15);
  int rb = (lane >> 4) << 3;
#pragma unroll
  for (int i = 0; i < 8; ++i) S[(rb + i) * ld + c] = (__bf16)vals[i];
}

__global__ void __launch_bounds__(WAVES * 32) painn_fused(
    const float* __restrict__ xs, const float* __restrict__ xp,
    const float* __restrict__ bU0, const float* __restrict__ bV0,
    const float* __restrict__ b1, const float* __restrict__ b2,
    const __bf16* __restrict__ W, float* __restrict__ out, int Nnodes) {
  extern __shared__ __align__(16) char dynsmem[];
  const int lane = threadIdx.x & 31;
  const int wv   = threadIdx.x >> 5;
  const int wid  = blockIdx.x * WAVES + wv;
  const int ntiles = Nnodes >> 4;
  if (wid >= ntiles) return;          // wave-uniform: EXEC stays all-ones for WMMA
  const int n0 = wid << 4;

  __bf16* sw     = (__bf16*)dynsmem + (size_t)wv * LDS_PER_WAVE;
  __bf16* sU     = sw;                // 30 tiles x 256
  __bf16* sDots  = sw + 7680;         // 16 x 224
  __bf16* sInv   = sw + 11264;        // 16 x 224
  __bf16* sH     = sw + 14848;        // 16 x 128
  __bf16* sInner = sw + 16896;        // 8 tiles x 256

  const __bf16* WU0t = W + OFF_WU0;
  const __bf16* WV0t = W + OFF_WV0;
  const __bf16* WU1t = W + OFF_WU1;
  const __bf16* WV1t = W + OFF_WV1;
  const __bf16* WU2t = W + OFF_WU2;
  const __bf16* WV2t = W + OFF_WV2;
  const __bf16* Wdt  = W + OFF_WDOT;
  const __bf16* Wm1t = W + OFF_WM1;
  const __bf16* Wm2t = W + OFF_WM2;

  const int cl = lane & 15;
  const int rb = (lane >> 4) << 3;

  // ---------------- Phase 0: l=0 (U0, V0, dot0, inv0) ----------------
  {
    v16bf ax[4];
#pragma unroll
    for (int ks = 0; ks < 4; ++ks) ax[ks] = loadA_g(xp, 480, n0, ks * 32, lane);
#pragma unroll
    for (int j = 0; j < 8; ++j) {
      v8f au = {}, av = {};
#pragma unroll
      for (int ks = 0; ks < 4; ++ks) {
        au = wmma_bf(ax[ks], loadB(WU0t, 128, j * 16, ks * 32, lane), au);
        av = wmma_bf(ax[ks], loadB(WV0t, 128, j * 16, ks * 32, lane), av);
      }
      float bu = bU0[j * 16 + cl], bv = bV0[j * 16 + cl];
      v8f d0v, i0v;
#pragma unroll
      for (int i = 0; i < 8; ++i) {
        float u = au[i] + bu, v = av[i] + bv;
        au[i]  = u;
        d0v[i] = u * v;
        i0v[i] = fabsf(v + E2V) - EPSV;
      }
      storeTile(sU + j * 256, au, lane);
      storeRowMajor(sDots, 224, j * 16, d0v, lane);
      storeRowMajor(sInv, 224, j * 16, i0v, lane);
    }
  }

  // ---------------- Phase 1: l=1 (U1, dot1, inv1) ----------------
  {
    v16bf a1[3][2];
#pragma unroll
    for (int m = 0; m < 3; ++m)
#pragma unroll
      for (int ks = 0; ks < 2; ++ks)
        a1[m][ks] = loadA_g_strided(xp, n0, 128, 3, m, ks * 32, lane);
#pragma unroll
    for (int vt = 0; vt < 4; ++vt) {
      v8f dacc = {}, vacc = {};
#pragma unroll
      for (int m = 0; m < 3; ++m) {
        v8f au = {}, av = {};
#pragma unroll
        for (int ks = 0; ks < 2; ++ks) {
          au = wmma_bf(a1[m][ks], loadB(WU1t, 64, vt * 16, ks * 32, lane), au);
          av = wmma_bf(a1[m][ks], loadB(WV1t, 64, vt * 16, ks * 32, lane), av);
        }
#pragma unroll
        for (int i = 0; i < 8; ++i) {
          dacc[i] += au[i] * av[i];
          float t = av[i] + E2V;
          vacc[i] += t * t;
        }
        storeTile(sU + (8 + vt * 3 + m) * 256, au, lane);
      }
      v8f invv;
#pragma unroll
      for (int i = 0; i < 8; ++i) invv[i] = sqrtf(vacc[i]) - EPSV;
      storeRowMajor(sDots, 224, 128 + vt * 16, dacc, lane);
      storeRowMajor(sInv, 224, 128 + vt * 16, invv, lane);
    }
  }

  // ---------------- Phase 2: l=2 (U2, dot2, inv2) ----------------
  {
    v16bf a2[5];
#pragma unroll
    for (int m = 0; m < 5; ++m) a2[m] = loadA_g_strided(xp, n0, 320, 5, m, 0, lane);
#pragma unroll
    for (int vt = 0; vt < 2; ++vt) {
      v8f dacc = {}, vacc = {};
#pragma unroll
      for (int m = 0; m < 5; ++m) {
        v8f au = {}, av = {};
        au = wmma_bf(a2[m], loadB(WU2t, 32, vt * 16, 0, lane), au);
        av = wmma_bf(a2[m], loadB(WV2t, 32, vt * 16, 0, lane), av);
#pragma unroll
        for (int i = 0; i < 8; ++i) {
          dacc[i] += au[i] * av[i];
          float t = av[i] + E2V;
          vacc[i] += t * t;
        }
        storeTile(sU + (20 + vt * 5 + m) * 256, au, lane);
      }
      v8f invv;
#pragma unroll
      for (int i = 0; i < 8; ++i) invv[i] = sqrtf(vacc[i]) - EPSV;
      storeRowMajor(sDots, 224, 192 + vt * 16, dacc, lane);
      storeRowMajor(sInv, 224, 192 + vt * 16, invv, lane);
    }
  }

  // ---------------- Phase 3: inner = dots @ W_dot (K=224) ----------------
#pragma unroll
  for (int j = 0; j < 8; ++j) {
    v8f acc = {};
#pragma unroll
    for (int ks = 0; ks < 7; ++ks)
      acc = wmma_bf(loadA_lds(sDots, 224, ks * 32, lane),
                    loadB(Wdt, 224, j * 16, ks * 32, lane), acc);
    storeTile(sInner + j * 256, acc, lane);
  }

  // ---------------- Phase 4: h = silu(mlp_in @ W_mlp1 + b1), K=352 ----------------
  {
    v16bf axs[4];
#pragma unroll
    for (int ks = 0; ks < 4; ++ks) axs[ks] = loadA_g(xs, 128, n0, ks * 32, lane);
#pragma unroll
    for (int j = 0; j < 8; ++j) {
      v8f acc = {};
#pragma unroll
      for (int ks = 0; ks < 4; ++ks)
        acc = wmma_bf(axs[ks], loadB(Wm1t, 352, j * 16, ks * 32, lane), acc);
#pragma unroll
      for (int ks = 4; ks < 11; ++ks)
        acc = wmma_bf(loadA_lds(sInv, 224, (ks - 4) * 32, lane),
                      loadB(Wm1t, 352, j * 16, ks * 32, lane), acc);
      float bb = b1[j * 16 + cl];
#pragma unroll
      for (int i = 0; i < 8; ++i) {
        float x = acc[i] + bb;
        acc[i]  = x / (1.f + __expf(-x));   // silu
      }
      storeRowMajor(sH, 128, j * 16, acc, lane);
    }
  }

  // ---------------- Phase 5: mlp_out = h @ W_mlp2 + b2, fused recombine ----------------
  v16bf ah[4];
#pragma unroll
  for (int ks = 0; ks < 4; ++ks) ah[ks] = loadA_lds(sH, 128, ks * 32, lane);

  auto gemm2 = [&](int j) -> v8f {
    v8f acc = {};
#pragma unroll
    for (int ks = 0; ks < 4; ++ks)
      acc = wmma_bf(ah[ks], loadB(Wm2t, 128, j * 16, ks * 32, lane), acc);
    float bb = b2[j * 16 + cl];
#pragma unroll
    for (int i = 0; i < 8; ++i) acc[i] += bb;
    return acc;
  };

  float* outC = out;
  float* outS = out + (size_t)Nnodes * 128;

  // a_vv (l=0): out_sph[:, 0:128]
#pragma unroll
  for (int j = 0; j < 8; ++j) {
    v8f avv = gemm2(j);
    float u[8];
    loadTile(sU + j * 256, u, lane);
    int c = j * 16 + cl;
#pragma unroll
    for (int i = 0; i < 8; ++i) {
      size_t idx = (size_t)(n0 + rb + i) * 480 + c;
      outS[idx] = xp[idx] + u[i] * avv[i];
    }
  }
  // a_vv (l=1): out_sph[:, 128 + v*3 + m]
#pragma unroll
  for (int vt = 0; vt < 4; ++vt) {
    v8f avv = gemm2(8 + vt);
    int v = vt * 16 + cl;
#pragma unroll
    for (int m = 0; m < 3; ++m) {
      float u[8];
      loadTile(sU + (8 + vt * 3 + m) * 256, u, lane);
      int c = 128 + v * 3 + m;
#pragma unroll
      for (int i = 0; i < 8; ++i) {
        size_t idx = (size_t)(n0 + rb + i) * 480 + c;
        outS[idx] = xp[idx] + u[i] * avv[i];
      }
    }
  }
  // a_vv (l=2): out_sph[:, 320 + v*5 + m]
#pragma unroll
  for (int vt = 0; vt < 2; ++vt) {
    v8f avv = gemm2(12 + vt);
    int v = vt * 16 + cl;
#pragma unroll
    for (int m = 0; m < 5; ++m) {
      float u[8];
      loadTile(sU + (20 + vt * 5 + m) * 256, u, lane);
      int c = 320 + v * 5 + m;
#pragma unroll
      for (int i = 0; i < 8; ++i) {
        size_t idx = (size_t)(n0 + rb + i) * 480 + c;
        outS[idx] = xp[idx] + u[i] * avv[i];
      }
    }
  }
  // a_sv / a_ss: out_scalar = x_scalar + a_sv * inner + a_ss
#pragma unroll
  for (int jj = 0; jj < 8; ++jj) {
    v8f asv = gemm2(14 + jj);
    v8f ass = gemm2(22 + jj);
    float inr[8];
    loadTile(sInner + jj * 256, inr, lane);
    int c = jj * 16 + cl;
#pragma unroll
    for (int i = 0; i < 8; ++i) {
      size_t idx = (size_t)(n0 + rb + i) * 128 + c;
      outC[idx] = xs[idx] + asv[i] * inr[i] + ass[i];
    }
  }
}

extern "C" void kernel_launch(void* const* d_in, const int* in_sizes, int n_in,
                              void* d_out, int out_size, void* d_ws, size_t ws_size,
                              hipStream_t stream) {
  const float* xs   = (const float*)d_in[0];
  const float* xp   = (const float*)d_in[1];
  const float* WU0  = (const float*)d_in[2];
  const float* WU1  = (const float*)d_in[3];
  const float* WU2  = (const float*)d_in[4];
  const float* bU0  = (const float*)d_in[5];
  const float* WV0  = (const float*)d_in[6];
  const float* WV1  = (const float*)d_in[7];
  const float* WV2  = (const float*)d_in[8];
  const float* bV0  = (const float*)d_in[9];
  const float* Wdot = (const float*)d_in[10];
  const float* Wm1  = (const float*)d_in[11];
  const float* b1   = (const float*)d_in[12];
  const float* Wm2  = (const float*)d_in[13];
  const float* b2   = (const float*)d_in[14];
  __bf16* W = (__bf16*)d_ws;

  int Nnodes = in_sizes[0] / 128;

  const float s128 = 0.08838834764831845f;  // 1/sqrt(128)
  const float s64  = 0.125f;                // 1/sqrt(64)
  const float s32  = 0.17677669529663687f;  // 1/sqrt(32)

  auto tp = [&](const float* src, int off, int K, int Nc, float sc) {
    int tot = K * Nc;
    prep_transpose<<<(tot + 255) / 256, 256, 0, stream>>>(src, W + off, K, Nc, sc);
  };
  tp(WU0, OFF_WU0, 128, 128, s128);
  tp(WV0, OFF_WV0, 128, 128, s128);
  tp(WU1, OFF_WU1, 64, 64, s64);
  tp(WV1, OFF_WV1, 64, 64, s64);
  tp(WU2, OFF_WU2, 32, 32, s32);
  tp(WV2, OFF_WV2, 32, 32, s32);
  tp(Wdot, OFF_WDOT, 224, 128, 1.0f);
  tp(Wm1, OFF_WM1, 352, 128, 1.0f);
  tp(Wm2, OFF_WM2, 128, 480, 1.0f);

  int ntiles = Nnodes / 16;
  int blocks = (ntiles + WAVES - 1) / WAVES;
  size_t smem = (size_t)WAVES * LDS_PER_WAVE * sizeof(__bf16);
  painn_fused<<<blocks, WAVES * 32, smem, stream>>>(xs, xp, bU0, bV0, b1, b2, W,
                                                    (float*)d_out, Nnodes);
}
